// Pointnet2Backbone_163208757635
// MI455X (gfx1250) — compile-verified, matrix operand order checked by reading, not run on hardware
//
#include <hip/hip_runtime.h>
#include <hip/hip_bf16.h>
#include <cstdint>
#include <cstddef>

// ---------------------------------------------------------------------------
// Types for CDNA5 WMMA (wave32): v_wmma_f32_16x16x32_f16
// ---------------------------------------------------------------------------
typedef __attribute__((ext_vector_type(16))) _Float16 v16h;
typedef __attribute__((ext_vector_type(8)))  _Float16 v8h;
typedef __attribute__((ext_vector_type(8)))  float    v8f;

#define TPB 256
#define NT  4   // n-tiles (16 cols each) per wave in the WMMA GEMM
static inline int ceil_div_i(long long a, long long b){ return (int)((a + b - 1) / b); }
static inline int rup32(int x){ return (x + 31) & ~31; }

// ---------------------------------------------------------------------------
// Split pointcloud (B,N,4) into xyz [B][N][3] and feat [B][1][N]
// ---------------------------------------------------------------------------
__global__ void k_extract(const float* __restrict__ pc, float* __restrict__ xyz,
                          float* __restrict__ feat, int B, int N){
  int i = blockIdx.x * blockDim.x + threadIdx.x;
  if (i >= B * N) return;
  int b = i / N, n = i % N;
  const float* p = pc + (size_t)i * 4;
  xyz[(size_t)i * 3 + 0] = p[0];
  xyz[(size_t)i * 3 + 1] = p[1];
  xyz[(size_t)i * 3 + 2] = p[2];
  feat[(size_t)b * N + n] = p[3];
}

// ---------------------------------------------------------------------------
// Farthest point sampling: one block per batch, LDS argmax reduction per step.
// out[it] = farthest index at entry of step it (matches the jax scan).
// ---------------------------------------------------------------------------
__global__ void k_fps(const float* __restrict__ xyz, float* __restrict__ dists,
                      int* __restrict__ inds, int N, int npoint){
  int b = blockIdx.x;
  const float* x = xyz + (size_t)b * N * 3;
  float* dd = dists + (size_t)b * N;
  int* out = inds + (size_t)b * npoint;
  __shared__ float sval[TPB];
  __shared__ int   sidx[TPB];
  __shared__ int   s_far;
  for (int n = threadIdx.x; n < N; n += blockDim.x) dd[n] = 1e10f;
  if (threadIdx.x == 0) s_far = 0;
  __syncthreads();
  for (int it = 0; it < npoint; ++it){
    int far = s_far;
    float cx = x[(size_t)far*3], cy = x[(size_t)far*3+1], cz = x[(size_t)far*3+2];
    float best = -1.0f; int besti = 0;
    for (int n = threadIdx.x; n < N; n += blockDim.x){
      float dx = x[(size_t)n*3] - cx, dy = x[(size_t)n*3+1] - cy, dz = x[(size_t)n*3+2] - cz;
      float d = dx*dx + dy*dy + dz*dz;
      float dm = fminf(dd[n], d);
      dd[n] = dm;
      if (dm > best){ best = dm; besti = n; }
    }
    sval[threadIdx.x] = best; sidx[threadIdx.x] = besti;
    __syncthreads();
    for (int s = TPB >> 1; s > 0; s >>= 1){
      if (threadIdx.x < s){
        if (sval[threadIdx.x + s] > sval[threadIdx.x] ||
            (sval[threadIdx.x + s] == sval[threadIdx.x] && sidx[threadIdx.x + s] < sidx[threadIdx.x])){
          sval[threadIdx.x] = sval[threadIdx.x + s];
          sidx[threadIdx.x] = sidx[threadIdx.x + s];
        }
      }
      __syncthreads();
    }
    if (threadIdx.x == 0){ out[it] = far; s_far = sidx[0]; }
    __syncthreads();
  }
}

__global__ void k_gather_xyz(const float* __restrict__ xyz, const int* __restrict__ inds,
                             float* __restrict__ nxyz, int N, int M, int B){
  int i = blockIdx.x * blockDim.x + threadIdx.x;
  if (i >= B * M) return;
  int b = i / M;
  int j = inds[i];
  const float* s = xyz + ((size_t)b * N + j) * 3;
  nxyz[(size_t)i*3+0] = s[0]; nxyz[(size_t)i*3+1] = s[1]; nxyz[(size_t)i*3+2] = s[2];
}

// ---------------------------------------------------------------------------
// Ball query: first K indices (ascending) inside radius, pad with first.
// ---------------------------------------------------------------------------
__global__ void k_ball(const float* __restrict__ xyz, const float* __restrict__ nxyz,
                       int* __restrict__ idx, int B, int N, int M, int K, float r2){
  int i = blockIdx.x * blockDim.x + threadIdx.x;
  if (i >= B * M) return;
  int b = i / M;
  const float* x = xyz + (size_t)b * N * 3;
  float cx = nxyz[(size_t)i*3], cy = nxyz[(size_t)i*3+1], cz = nxyz[(size_t)i*3+2];
  int* o = idx + (size_t)i * K;
  int cnt = 0;
  for (int n = 0; n < N && cnt < K; ++n){
    float dx = x[(size_t)n*3] - cx, dy = x[(size_t)n*3+1] - cy, dz = x[(size_t)n*3+2] - cz;
    if (dx*dx + dy*dy + dz*dz < r2) o[cnt++] = n;
  }
  int f = (cnt > 0) ? o[0] : (N - 1);
  for (; cnt < K; ++cnt) o[cnt] = f;
}

// ---------------------------------------------------------------------------
// Build f16 GEMM input rows [Ntot][Cpad]: c<3 -> (xyz - center)/r, then feats.
// ---------------------------------------------------------------------------
__global__ void k_group(const float* __restrict__ xyz, const float* __restrict__ nxyz,
                        const float* __restrict__ feat, const int* __restrict__ idx,
                        _Float16* __restrict__ IN, int B, int N, int M, int K,
                        int Cf, int Cpad, float inv_r, long long total){
  long long i = (long long)blockIdx.x * blockDim.x + threadIdx.x;
  if (i >= total) return;
  int c = (int)(i % Cpad);
  long long r = i / Cpad;          // r = (b*M+m)*K + k
  long long bm = r / K;            // b*M + m
  int b = (int)(bm / M);
  int j = idx[r];
  float v = 0.0f;
  if (c < 3){
    v = (xyz[((size_t)b * N + j) * 3 + c] - nxyz[(size_t)bm * 3 + c]) * inv_r;
  } else if (c < 3 + Cf){
    v = feat[((size_t)b * Cf + (c - 3)) * N + j];
  }
  IN[i] = (_Float16)v;
}

// Convert weight (cout,cin) f32 -> f16 padded (cout, cpad)
__global__ void k_wconv(const float* __restrict__ W, _Float16* __restrict__ Wh,
                        int cout, int cin, int cpad){
  int i = blockIdx.x * blockDim.x + threadIdx.x;
  if (i >= cout * cpad) return;
  int c = i % cpad, o = i / cpad;
  Wh[i] = (_Float16)(c < cin ? W[(size_t)o * cin + c] : 0.0f);
}

// ---------------------------------------------------------------------------
// WMMA GEMM (register-blocked): OUT[o][n] = sum_c W[o][c] * IN[n][c]
//   Wt : f16 [Cout][Cpad]   (A matrix tiles, 16x32)
//   IN : f16 [Ntot][Cpad]   (B matrix tiles, 32x16, rows = n columns)
//   OUT: f32 [Cout][Ntot]
// Each wave computes NT=4 n-tiles (64 columns) of one 16-row o-tile.
// Per K-step: ONE A load shared by 4 WMMAs. An empty inline-asm scheduling
// barrier pins all four B fragments live in DISTINCT register tuples and
// prevents the scheduler from sinking loads below it -> all ~10 b128 loads
// issue back-to-back (max MLP), one wait, then 4 back-to-back WMMAs.
// Per-lane layout follows CDNA5 ISA 7.12.2 (lane = half*16 + m; element e
// covers K = {half*8 + e (e<8), 16 + half*8 + (e-8)} within a k0..k0+31 step).
// ---------------------------------------------------------------------------
__global__ void __launch_bounds__(TPB)
k_gemm(const _Float16* __restrict__ Wt, const _Float16* __restrict__ IN,
       float* __restrict__ OUT, int Cpad, long long Ntot){
  int lane = threadIdx.x & 31;
  int wv   = threadIdx.x >> 5;
  long long n0 = ((long long)blockIdx.y * 8 + wv) * (16 * NT);
  if (n0 >= Ntot) return;                 // whole-wave uniform exit
  int o0 = blockIdx.x * 16;
  int m    = lane & 15;
  int half = lane >> 4;

  const _Float16* wrow = Wt + (size_t)(o0 + m) * Cpad + half * 8;
  const _Float16* irow[NT];
  #pragma unroll
  for (int t = 0; t < NT; ++t)
    irow[t] = IN + (size_t)(n0 + t * 16 + m) * Cpad + half * 8;

  v8f acc[NT];
  #pragma unroll
  for (int t = 0; t < NT; ++t) acc[t] = (v8f){0.f,0.f,0.f,0.f,0.f,0.f,0.f,0.f};

  union pk { v16h v; v8h h[2]; };

  for (int k0 = 0; k0 < Cpad; k0 += 32){
    // ---- load phase: A once + all NT B tiles ----
    pk ua;
    ua.h[0] = *(const v8h*)(wrow + k0);
    ua.h[1] = *(const v8h*)(wrow + k0 + 16);
    pk ub[NT];
    #pragma unroll
    for (int t = 0; t < NT; ++t){
      __builtin_prefetch((const void*)(irow[t] + k0 + 32), 0, 1);  // global_prefetch_b8
      ub[t].h[0] = *(const v8h*)(irow[t] + k0);
      ub[t].h[1] = *(const v8h*)(irow[t] + k0 + 16);
    }
    // ---- scheduling barrier: keep all fragments live in distinct regs,
    //      forbid sinking any load below this point ----
    asm volatile("" :
        "+v"(ua.h[0]),    "+v"(ua.h[1]),
        "+v"(ub[0].h[0]), "+v"(ub[0].h[1]),
        "+v"(ub[1].h[0]), "+v"(ub[1].h[1]),
        "+v"(ub[2].h[0]), "+v"(ub[2].h[1]),
        "+v"(ub[3].h[0]), "+v"(ub[3].h[1]));
    // ---- compute phase: 4 independent WMMA chains, no waits between ----
    #pragma unroll
    for (int t = 0; t < NT; ++t){
      acc[t] = __builtin_amdgcn_wmma_f32_16x16x32_f16(
          /*neg_a=*/false, ua.v, /*neg_b=*/false, ub[t].v,
          /*c_mod=*/(short)0, acc[t], /*reuse_a=*/false, /*reuse_b=*/false);
    }
  }
  #pragma unroll
  for (int t = 0; t < NT; ++t){
    long long n = n0 + t * 16 + m;
    #pragma unroll
    for (int v = 0; v < 8; ++v){
      int o = o0 + half * 8 + v;         // D layout: VGPR v holds M=v (+8 for upper lanes)
      OUT[(size_t)o * Ntot + n] = acc[t][v];
    }
  }
}

// ---------------------------------------------------------------------------
// BN stats: one block per channel; population mean/var over Ntot.
// ---------------------------------------------------------------------------
__global__ void k_bnstats(const float* __restrict__ OUT, long long Ntot,
                          float* __restrict__ mean, float* __restrict__ var){
  int c = blockIdx.x;
  const float* row = OUT + (size_t)c * Ntot;
  __shared__ float ss[TPB], s2[TPB];
  float a = 0.f, q = 0.f;
  for (long long n = threadIdx.x; n < Ntot; n += blockDim.x){
    float x = row[n]; a += x; q += x * x;
  }
  ss[threadIdx.x] = a; s2[threadIdx.x] = q;
  __syncthreads();
  for (int s = TPB >> 1; s > 0; s >>= 1){
    if (threadIdx.x < s){ ss[threadIdx.x] += ss[threadIdx.x+s]; s2[threadIdx.x] += s2[threadIdx.x+s]; }
    __syncthreads();
  }
  if (threadIdx.x == 0){
    float mm = ss[0] / (float)Ntot;
    mean[c] = mm;
    var[c]  = s2[0] / (float)Ntot - mm * mm;
  }
}

// BN + ReLU; write normalized f32 back and f16 next-layer input [Ntot][Cnext]
__global__ void k_bnapply_next(float* __restrict__ OUT, const float* __restrict__ mean,
                               const float* __restrict__ var, const float* __restrict__ g,
                               const float* __restrict__ bb, int Cout, long long Ntot,
                               _Float16* __restrict__ NEXT, int Cnext, long long total){
  long long i = (long long)blockIdx.x * blockDim.x + threadIdx.x;
  if (i >= total) return;
  int c = (int)(i % Cnext);
  long long n = i / Cnext;
  if (c < Cout){
    float x = OUT[(size_t)c * Ntot + n];
    float y = (x - mean[c]) * rsqrtf(var[c] + 1e-5f) * g[c] + bb[c];
    y = fmaxf(y, 0.0f);
    OUT[(size_t)c * Ntot + n] = y;
    NEXT[i] = (_Float16)y;
  } else {
    NEXT[i] = (_Float16)0.0f;
  }
}

__global__ void k_bnapply_final(float* __restrict__ OUT, const float* __restrict__ mean,
                                const float* __restrict__ var, const float* __restrict__ g,
                                const float* __restrict__ bb, int Cout, long long Ntot){
  long long i = (long long)blockIdx.x * blockDim.x + threadIdx.x;
  if (i >= (long long)Cout * Ntot) return;
  int c = (int)(i / Ntot);
  float x = OUT[i];
  float y = (x - mean[c]) * rsqrtf(var[c] + 1e-5f) * g[c] + bb[c];
  OUT[i] = fmaxf(y, 0.0f);
}

// Max over nsample: nfeat[b][c][m] = max_k OUT[c][(b*M+m)*K + k]
__global__ void k_maxpool(const float* __restrict__ OUT, float* __restrict__ nfeat,
                          int B, int M, int K, int Cout, long long Ntot){
  int i = blockIdx.x * blockDim.x + threadIdx.x;
  if (i >= B * Cout * M) return;
  int b = i / (Cout * M);
  int rem = i % (Cout * M);
  int c = rem / M, m = rem % M;
  long long base = ((long long)b * M + m) * K;
  const float* row = OUT + (size_t)c * Ntot + base;
  float mx = -3.0e38f;
  for (int k = 0; k < K; ++k) mx = fmaxf(mx, row[k]);
  nfeat[i] = mx;
}

// 3-NN (squared dists) + inverse-distance weights
__global__ void k_3nn(const float* __restrict__ uxyz, const float* __restrict__ kxyz,
                      int B, int Mu, int Mk, int* __restrict__ idx3, float* __restrict__ w3){
  int i = blockIdx.x * blockDim.x + threadIdx.x;
  if (i >= B * Mu) return;
  int b = i / Mu;
  float px = uxyz[(size_t)i*3], py = uxyz[(size_t)i*3+1], pz = uxyz[(size_t)i*3+2];
  const float* kx = kxyz + (size_t)b * Mk * 3;
  float d0 = 3e38f, d1 = 3e38f, d2 = 3e38f; int i0 = 0, i1 = 0, i2 = 0;
  for (int j = 0; j < Mk; ++j){
    float dx = kx[(size_t)j*3] - px, dy = kx[(size_t)j*3+1] - py, dz = kx[(size_t)j*3+2] - pz;
    float d = dx*dx + dy*dy + dz*dz;
    if (d < d0){ d2=d1;i2=i1; d1=d0;i1=i0; d0=d;i0=j; }
    else if (d < d1){ d2=d1;i2=i1; d1=d;i1=j; }
    else if (d < d2){ d2=d;i2=j; }
  }
  float r0 = 1.0f/(d0+1e-8f), r1 = 1.0f/(d1+1e-8f), r2 = 1.0f/(d2+1e-8f);
  float s = 1.0f/(r0+r1+r2);
  idx3[(size_t)i*3] = i0; idx3[(size_t)i*3+1] = i1; idx3[(size_t)i*3+2] = i2;
  w3[(size_t)i*3] = r0*s; w3[(size_t)i*3+1] = r1*s; w3[(size_t)i*3+2] = r2*s;
}

// FP input rows [B*Mu][2C]: c<C -> interpolated known feats, else unknown feats
__global__ void k_fpin(const float* __restrict__ kfeat, const float* __restrict__ ufeat,
                       const int* __restrict__ idx3, const float* __restrict__ w3,
                       _Float16* __restrict__ IN, int B, int Mu, int Mk, int C,
                       long long total){
  long long i = (long long)blockIdx.x * blockDim.x + threadIdx.x;
  if (i >= total) return;
  int c = (int)(i % (2 * C));
  long long r = i / (2 * C);       // b*Mu + u
  int u = (int)(r % Mu);
  int b = (int)(r / Mu);
  float v;
  if (c < C){
    const float* kf = kfeat + ((size_t)b * C + c) * Mk;
    const int* id = idx3 + (size_t)r * 3;
    const float* w = w3 + (size_t)r * 3;
    v = w[0]*kf[id[0]] + w[1]*kf[id[1]] + w[2]*kf[id[2]];
  } else {
    v = ufeat[((size_t)b * C + (c - C)) * Mu + u];
  }
  IN[i] = (_Float16)v;
}

// OUT [C][B*Mu] -> dst [B][C][Mu]
__global__ void k_tr(const float* __restrict__ OUT, float* __restrict__ dst,
                     int B, int C, int Mu){
  int i = blockIdx.x * blockDim.x + threadIdx.x;
  if (i >= B * C * Mu) return;
  int b = i / (C * Mu);
  int rem = i % (C * Mu);
  int c = rem / Mu, u = rem % Mu;
  dst[i] = OUT[(size_t)c * ((size_t)B * Mu) + (size_t)b * Mu + u];
}

// ---------------------------------------------------------------------------
// Host-side orchestration
// ---------------------------------------------------------------------------
struct Layer { const float* W; const float* g; const float* b; int cin; int cout; };

static void run_mlp(const Layer* Ls, int nL, _Float16* INa, _Float16* INb,
                    float* OUT, _Float16* Wh, float* mean, float* var,
                    long long Ntot, hipStream_t stream){
  _Float16* cur = INa; _Float16* nxt = INb;
  for (int l = 0; l < nL; ++l){
    int cin = Ls[l].cin, cout = Ls[l].cout, cpad = rup32(cin);
    k_wconv<<<ceil_div_i((long long)cout * cpad, TPB), TPB, 0, stream>>>(Ls[l].W, Wh, cout, cin, cpad);
    dim3 g((unsigned)(cout / 16), (unsigned)ceil_div_i(Ntot, 16 * 8 * NT));
    k_gemm<<<g, TPB, 0, stream>>>(Wh, cur, OUT, cpad, Ntot);
    k_bnstats<<<cout, TPB, 0, stream>>>(OUT, Ntot, mean, var);
    if (l + 1 < nL){
      int cnext = rup32(cout);
      long long total = Ntot * cnext;
      k_bnapply_next<<<ceil_div_i(total, TPB), TPB, 0, stream>>>(
          OUT, mean, var, Ls[l].g, Ls[l].b, cout, Ntot, nxt, cnext, total);
      _Float16* t = cur; cur = nxt; nxt = t;
    } else {
      k_bnapply_final<<<ceil_div_i((long long)cout * Ntot, TPB), TPB, 0, stream>>>(
          OUT, mean, var, Ls[l].g, Ls[l].b, cout, Ntot);
    }
  }
}

extern "C" void kernel_launch(void* const* d_in, const int* in_sizes, int n_in,
                              void* d_out, int out_size, void* d_ws, size_t ws_size,
                              hipStream_t stream){
  (void)in_sizes; (void)n_in; (void)out_size; (void)ws_size;
  const int B = 4, N0 = 20000;
  const float* pc = (const float*)d_in[0];

  // ---- workspace carve (256B aligned) ----
  size_t off = 0;
  auto carve = [&](size_t bytes) -> void* {
    void* p = (char*)d_ws + off;
    off = (off + bytes + 255) & ~(size_t)255;
    return p;
  };
  float* xyz0   = (float*)carve((size_t)B * N0 * 3 * 4);
  float* feat0  = (float*)carve((size_t)B * N0 * 4);
  float* dists  = (float*)carve((size_t)B * N0 * 4);
  int*   inds   = (int*)  carve((size_t)B * 2048 * 4);
  float* sa1_xyz = (float*)carve((size_t)B * 2048 * 3 * 4);
  float* sa2_xyz = (float*)carve((size_t)B * 1024 * 3 * 4);
  float* sa3_xyz = (float*)carve((size_t)B * 512  * 3 * 4);
  float* sa4_xyz = (float*)carve((size_t)B * 256  * 3 * 4);
  float* sa1_f = (float*)carve((size_t)B * 128 * 2048 * 4);
  float* sa2_f = (float*)carve((size_t)B * 256 * 1024 * 4);
  float* sa3_f = (float*)carve((size_t)B * 256 * 512  * 4);
  float* sa4_f = (float*)carve((size_t)B * 256 * 256  * 4);
  int*   ballidx = (int*)carve((size_t)B * 2048 * 64 * 4);
  _Float16* INa = (_Float16*)carve((size_t)524288 * 64 * 2);   // 64 MB
  _Float16* INb = (_Float16*)carve((size_t)524288 * 64 * 2);   // 64 MB
  float* OUT    = (float*)carve((size_t)128 * 524288 * 4);     // 256 MB
  _Float16* Wh  = (_Float16*)carve((size_t)256 * 512 * 2);
  float* mean   = (float*)carve(256 * 4);
  float* var    = (float*)carve(256 * 4);
  int*   idx3   = (int*)  carve((size_t)B * 1024 * 3 * 4);
  float* w3     = (float*)carve((size_t)B * 1024 * 3 * 4);
  float* fp1_out = (float*)carve((size_t)B * 256 * 512 * 4);

  // ---- layer tables (d_in order: pc, sa1..sa4 (3 layers x W,g,b), fp1, fp2) ----
  auto L = [&](int base, int layer, int cin, int cout) -> Layer {
    return Layer{ (const float*)d_in[base + layer*3 + 0],
                  (const float*)d_in[base + layer*3 + 1],
                  (const float*)d_in[base + layer*3 + 2], cin, cout };
  };
  Layer sa1L[3] = { L(1,0,4,64),    L(1,1,64,64),   L(1,2,64,128)  };
  Layer sa2L[3] = { L(10,0,131,128),L(10,1,128,128),L(10,2,128,256)};
  Layer sa3L[3] = { L(19,0,259,128),L(19,1,128,128),L(19,2,128,256)};
  Layer sa4L[3] = { L(28,0,259,128),L(28,1,128,128),L(28,2,128,256)};
  Layer fp1L[2] = { L(37,0,512,256),L(37,1,256,256) };
  Layer fp2L[2] = { L(43,0,512,256),L(43,1,256,256) };

  k_extract<<<ceil_div_i((long long)B * N0, TPB), TPB, 0, stream>>>(pc, xyz0, feat0, B, N0);

  auto sa_module = [&](const float* xyz_in, int Nin, const float* feat_in, int Cf,
                       int M, int K, float radius, const Layer* Ls,
                       float* nxyz, float* nfeat){
    k_fps<<<B, TPB, 0, stream>>>(xyz_in, dists, inds, Nin, M);
    k_gather_xyz<<<ceil_div_i((long long)B * M, TPB), TPB, 0, stream>>>(xyz_in, inds, nxyz, Nin, M, B);
    k_ball<<<ceil_div_i((long long)B * M, TPB), TPB, 0, stream>>>(
        xyz_in, nxyz, ballidx, B, Nin, M, K, radius * radius);
    long long Ntot = (long long)B * M * K;
    int Cpad0 = rup32(3 + Cf);
    long long total = Ntot * Cpad0;
    k_group<<<ceil_div_i(total, TPB), TPB, 0, stream>>>(
        xyz_in, nxyz, feat_in, ballidx, INa, B, Nin, M, K, Cf, Cpad0, 1.0f / radius, total);
    run_mlp(Ls, 3, INa, INb, OUT, Wh, mean, var, Ntot, stream);
    int coutL = Ls[2].cout;
    k_maxpool<<<ceil_div_i((long long)B * coutL * M, TPB), TPB, 0, stream>>>(
        OUT, nfeat, B, M, K, coutL, Ntot);
  };

  auto fp_module = [&](const float* uxyz, int Mu, const float* kxyz, int Mk,
                       const float* ufeat, const float* kfeat, int C,
                       const Layer* Ls, float* dst){
    k_3nn<<<ceil_div_i((long long)B * Mu, TPB), TPB, 0, stream>>>(uxyz, kxyz, B, Mu, Mk, idx3, w3);
    long long Ntot = (long long)B * Mu;
    long long total = Ntot * 2 * C;
    k_fpin<<<ceil_div_i(total, TPB), TPB, 0, stream>>>(
        kfeat, ufeat, idx3, w3, INa, B, Mu, Mk, C, total);
    run_mlp(Ls, 2, INa, INb, OUT, Wh, mean, var, Ntot, stream);
    k_tr<<<ceil_div_i((long long)B * Ls[1].cout * Mu, TPB), TPB, 0, stream>>>(
        OUT, dst, B, Ls[1].cout, Mu);
  };

  sa_module(xyz0,    N0,   feat0, 1,   2048, 64, 0.2f, sa1L, sa1_xyz, sa1_f);
  sa_module(sa1_xyz, 2048, sa1_f, 128, 1024, 32, 0.4f, sa2L, sa2_xyz, sa2_f);
  sa_module(sa2_xyz, 1024, sa2_f, 256, 512,  16, 0.8f, sa3L, sa3_xyz, sa3_f);
  sa_module(sa3_xyz, 512,  sa3_f, 256, 256,  16, 1.2f, sa4L, sa4_xyz, sa4_f);

  fp_module(sa3_xyz, 512,  sa4_xyz, 256, sa3_f, sa4_f,   256, fp1L, fp1_out);
  fp_module(sa2_xyz, 1024, sa3_xyz, 512, sa2_f, fp1_out, 256, fp2L, (float*)d_out);
}
